// XANES_E3GNN_20306605376047
// MI455X (gfx1250) — matrix operand order repaired and checked by reading, not compiled, not run on hardware
//
#include <hip/hip_runtime.h>
#include <hip/hip_bf16.h>
#include <math.h>

typedef __attribute__((ext_vector_type(16))) _Float16 v16h;
typedef __attribute__((ext_vector_type(8)))  _Float16 v8h;
typedef __attribute__((ext_vector_type(8)))  float    v8f;

#define MUL0V 64
#define MUL1V 32
#define MUL2V 16
#define NRAD 10
#define NLAYERS_C 4
#define RF_N 112
#define PER_B 50
#define R_MAXF 5.0f

struct __align__(8) H4 { _Float16 a, b, c, d; };

__device__ __forceinline__ float silu_f(float x) { return x / (1.f + expf(-x)); }

// ---- WMMA fragment helpers (16x16x32 f16, wave32) ------------------------
// A 16x32 f16: lane l holds row m=l&15; group g=l>>4; halves [0..15] are
// K=16g..16g+15 -> one contiguous 32B run.
__device__ __forceinline__ v16h load_a_ptr(const _Float16* rowPtr, int kt, int g) {
  union { v16h v; v8h h[2]; } u;
  const _Float16* p = rowPtr + kt * 32 + 16 * g;
  u.h[0] = *(const v8h*)(p);
  u.h[1] = *(const v8h*)(p + 8);
  return u.v;
}
// B 32x16 f16 packed tiles: element (k,n): lane = n + 16*(k>=16), half = k&15.
// Packed so each lane reads 16 contiguous halves (32B).
__device__ __forceinline__ v16h load_b_frag(const _Float16* pack, int kt, int nt, int NT, int lane) {
  const _Float16* p = pack + (((size_t)(kt * NT + nt)) * 32 + lane) * 16;
  union { v16h v; v8h h[2]; } u;
  u.h[0] = *(const v8h*)(p);
  u.h[1] = *(const v8h*)(p + 8);
  return u.v;
}
__device__ __forceinline__ v8f wmma16(v16h a, v16h b, v8f c) {
  return __builtin_amdgcn_wmma_f32_16x16x32_f16(false, a, false, b, (short)0, c, false, false);
}
__device__ __forceinline__ void wave_lds_fence() {
  __builtin_amdgcn_wave_barrier();
  asm volatile("s_wait_dscnt 0x0" ::: "memory");
  __builtin_amdgcn_wave_barrier();
}

// ---- pack f32 weight (Kreal x Ncols, row stride ld) into B-fragment order --
__global__ void pack_b_kernel(const float* __restrict__ src, int Kreal, int Ncols, int ld,
                              _Float16* __restrict__ dst, int NT, int ntOff, int KT) {
  int idx = blockIdx.x * blockDim.x + threadIdx.x;
  int total = KT * 32 * Ncols;
  if (idx >= total) return;
  int k = idx / Ncols, n = idx % Ncols;
  float v = (k < Kreal) ? src[(size_t)k * ld + n] : 0.f;
  int ac = ntOff * 16 + n;
  int nt = ac >> 4, nn = ac & 15;
  int kt = k >> 5, kk = k & 31;
  int lane = nn + ((kk >= 16) ? 16 : 0);
  int half = kk & 15;
  dst[(((size_t)(kt * NT + nt)) * 32 + lane) * 16 + half] = (_Float16)v;
}

__global__ void zero_kernel(float* p, long long count) {
  long long i = (long long)blockIdx.x * blockDim.x + threadIdx.x;
  if (i < count) p[i] = 0.f;
}

// ---- per-edge geometry: sh (E x 9 f32), rbas padded to K=32 f16 ------------
__global__ void edge_geom_kernel(const int* __restrict__ ei, const float* __restrict__ pos,
                                 const float* __restrict__ shift,
                                 float* __restrict__ sh, _Float16* __restrict__ rbas, int E) {
  int e = blockIdx.x * blockDim.x + threadIdx.x;
  if (e >= E) return;
  int s = ei[e], d = ei[E + e];
  float vx = pos[d * 3 + 0] - pos[s * 3 + 0] + shift[e * 3 + 0];
  float vy = pos[d * 3 + 1] - pos[s * 3 + 1] + shift[e * 3 + 1];
  float vz = pos[d * 3 + 2] - pos[s * 3 + 2] + shift[e * 3 + 2];
  float r = sqrtf(vx * vx + vy * vy + vz * vz);
  float inv = 1.f / (r + 1e-9f);
  float x = vx * inv, y = vy * inv, z = vz * inv;
  const float c3 = 1.7320508075688772f, c15 = 3.872983346207417f, c5 = 2.23606797749979f;
  float* o = sh + (size_t)e * 9;
  o[0] = 1.f; o[1] = c3 * x; o[2] = c3 * y; o[3] = c3 * z;
  o[4] = c15 * x * y; o[5] = c15 * y * z; o[6] = 0.5f * c5 * (3.f * z * z - 1.f);
  o[7] = c15 * x * z; o[8] = 0.5f * c15 * (x * x - y * y);
  const float gamma = R_MAXF / (NRAD - 1);
  _Float16* rb = rbas + (size_t)e * 32;
  for (int j = 0; j < 32; ++j) {
    float v = 0.f;
    if (j < NRAD) { float t = (r - (float)j * gamma) / gamma; v = expf(-t * t); }
    rb[j] = (_Float16)v;
  }
}

__global__ void init_s_kernel(const int* __restrict__ z, const float* __restrict__ emb,
                              float* __restrict__ sbuf, int N) {
  int i = blockIdx.x * blockDim.x + threadIdx.x;
  if (i >= N * 64) return;
  int n = i >> 6, c = i & 63;
  sbuf[i] = emb[z[n] * 64 + c];
}

// ---- fused per-edge kernel: radial MLP (2 GEMMs) + message GEMM + scatter --
__global__ void __launch_bounds__(256)
edge_msg_kernel(const int* __restrict__ src, const int* __restrict__ dst,
                const _Float16* __restrict__ rbas, const float* __restrict__ sh,
                const float* __restrict__ s_in,
                const _Float16* __restrict__ rw1p, const float* __restrict__ rb1,
                const _Float16* __restrict__ rw2p, const float* __restrict__ rb2,
                const _Float16* __restrict__ wpack,
                float* __restrict__ a0, float* __restrict__ av, float* __restrict__ at, int E) {
  const int w = threadIdx.x >> 5, lane = threadIdx.x & 31;
  const int m = lane & 15, g16 = lane >> 4;
  __shared__ __align__(16) _Float16 lds_ss[8][16][64];
  __shared__ __align__(16) _Float16 lds_hid[8][16][64];
  int nT = (E + 15) / 16;
  int tile = blockIdx.x * 8 + w;
  bool valid_tile = tile < nT;
  int tIdx = valid_tile ? tile : (nT - 1);
  int e0 = tIdx * 16;

  // stage ss = f16(s[src[e]]) rows into LDS (lane covers row m, cols 32*g16..)
  {
    int e = e0 + m; if (e >= E) e = E - 1;
    int sn = src[e];
    const float4* ps = (const float4*)(s_in + (size_t)sn * 64 + g16 * 32);
#pragma unroll
    for (int j = 0; j < 8; ++j) {
      float4 x = ps[j];
      H4 h; h.a = (_Float16)x.x; h.b = (_Float16)x.y; h.c = (_Float16)x.z; h.d = (_Float16)x.w;
      *(H4*)&lds_ss[w][m][g16 * 32 + j * 4] = h;
    }
  }
  // A fragment of rbas directly from pre-padded global rows (K=32)
  int eA = e0 + m; if (eA >= E) eA = E - 1;
  v16h a_rb = load_a_ptr(rbas + (size_t)eA * 32, 0, g16);

  // rf1 = silu(rbas @ rw1 + rb1)  -> LDS hidden (16x64 f16)
#pragma unroll
  for (int nt = 0; nt < 4; ++nt) {
    v16h b = load_b_frag(rw1p, 0, nt, 4, lane);
    v8f acc = {};
    acc = wmma16(a_rb, b, acc);
    int n = nt * 16 + m;
    float bi = rb1[n];
#pragma unroll
    for (int gg = 0; gg < 8; ++gg) {
      int row = gg + 8 * g16;
      lds_hid[w][row][n] = (_Float16)silu_f(acc[gg] + bi);
    }
  }
  wave_lds_fence();

  v16h a_h0 = load_a_ptr(&lds_hid[w][m][0], 0, g16);
  v16h a_h1 = load_a_ptr(&lds_hid[w][m][0], 1, g16);
  v16h a_s0 = load_a_ptr(&lds_ss[w][m][0], 0, g16);
  v16h a_s1 = load_a_ptr(&lds_ss[w][m][0], 1, g16);

  int drow[8]; int erow[8];
#pragma unroll
  for (int gg = 0; gg < 8; ++gg) {
    int e = e0 + gg + 8 * g16; if (e >= E) e = E - 1;
    erow[gg] = e; drow[gg] = dst[e];
  }
  const float invDeg = 0.25f;  // 1/sqrt(E/N) = 1/4

  for (int nt = 0; nt < 7; ++nt) {
    v8f aR = {};
    aR = wmma16(a_h0, load_b_frag(rw2p, 0, nt, 7, lane), aR);
    aR = wmma16(a_h1, load_b_frag(rw2p, 1, nt, 7, lane), aR);
    v8f aM = {};
    aM = wmma16(a_s0, load_b_frag(wpack, 0, nt, 7, lane), aM);
    aM = wmma16(a_s1, load_b_frag(wpack, 1, nt, 7, lane), aM);
    int n = nt * 16 + m;
    float b2 = rb2[n];
#pragma unroll
    for (int gg = 0; gg < 8; ++gg) {
      int e = e0 + gg + 8 * g16;
      if (!valid_tile || e >= E) continue;
      float val = aM[gg] * (aR[gg] + b2) * invDeg;
      int d = drow[gg];
      if (n < 64) {
        atomicAdd(&a0[(size_t)d * 64 + n], val);
      } else if (n < 96) {
        int ch = n - 64;
        const float* se = sh + (size_t)erow[gg] * 9;
        atomicAdd(&av[((size_t)d * 32 + ch) * 3 + 0], val * se[1]);
        atomicAdd(&av[((size_t)d * 32 + ch) * 3 + 1], val * se[2]);
        atomicAdd(&av[((size_t)d * 32 + ch) * 3 + 2], val * se[3]);
      } else {
        int ch = n - 96;
        const float* se = sh + (size_t)erow[gg] * 9;
#pragma unroll
        for (int c = 0; c < 5; ++c)
          atomicAdd(&at[((size_t)d * 16 + ch) * 5 + c], val * se[4 + c]);
      }
    }
  }
}

// ---- generic WMMA GEMM: C = act( A[rowIdx] @ Bpack + bias + addend ) -------
__global__ void __launch_bounds__(256)
wmma_gemm_kernel(const float* __restrict__ A, const int* __restrict__ rowIdx,
                 int M, int Kreal, int ldA,
                 const _Float16* __restrict__ Bpack, int KT, int NT,
                 const float* __restrict__ bias, const float* __restrict__ addend,
                 int act, float* __restrict__ Cout, int Nout) {
  const int w = threadIdx.x >> 5, lane = threadIdx.x & 31;
  const int m = lane & 15, g16 = lane >> 4;
  __shared__ __align__(16) _Float16 lds[8][16][192];
  int nTiles = (M + 15) / 16;
  int tile = blockIdx.x * 8 + w;
  bool active = tile < nTiles;
  int tIdx = active ? tile : (nTiles - 1);
  int r = tIdx * 16 + m;
  int rl = (r < M) ? r : (M - 1);
  int rowA = rowIdx ? rowIdx[rl] : rl;
  for (int c = g16; c < KT * 32; c += 2) {
    float x = (c < Kreal) ? A[(size_t)rowA * ldA + c] : 0.f;
    lds[w][m][c] = (_Float16)x;
  }
  wave_lds_fence();
  for (int nt = 0; nt < NT; ++nt) {
    v8f acc = {};
    for (int kt = 0; kt < KT; ++kt) {
      v16h a = load_a_ptr(&lds[w][m][0], kt, g16);
      v16h b = load_b_frag(Bpack, kt, nt, NT, lane);
      acc = wmma16(a, b, acc);
    }
    int n = nt * 16 + m;
    float bi = bias ? bias[n] : 0.f;
#pragma unroll
    for (int gg = 0; gg < 8; ++gg) {
      int rr = tIdx * 16 + gg + 8 * g16;
      if (!active || rr >= M) continue;
      float v = acc[gg] + bi + (addend ? addend[(size_t)rr * Nout + n] : 0.f);
      if (act) v = silu_f(v);
      Cout[(size_t)rr * Nout + n] = v;
    }
  }
}

// ---- v/t channel mixing (small, VALU) --------------------------------------
__global__ void vt_update_kernel(const float* __restrict__ vin, const float* __restrict__ tin,
                                 const float* __restrict__ s1, const float* __restrict__ s2,
                                 const float* __restrict__ av, const float* __restrict__ at,
                                 float* __restrict__ vout, float* __restrict__ tout, int N) {
  long long idx = (long long)blockIdx.x * blockDim.x + threadIdx.x;
  long long nv = (long long)N * 96;
  if (idx < nv) {
    int n = (int)(idx / 96); int rem = (int)(idx % 96); int k = rem / 3, c = rem % 3;
    float acc = av[idx];
    for (int mm = 0; mm < 32; ++mm) acc += vin[((size_t)n * 32 + mm) * 3 + c] * s1[mm * 32 + k];
    vout[idx] = acc;
  } else {
    long long j = idx - nv;
    if (j >= (long long)N * 80) return;
    int n = (int)(j / 80); int rem = (int)(j % 80); int k = rem / 5, c = rem % 5;
    float acc = at[j];
    for (int mm = 0; mm < 16; ++mm) acc += tin[((size_t)n * 16 + mm) * 5 + c] * s2[mm * 16 + k];
    tout[j] = acc;
  }
}

__global__ void absidx_kernel(const unsigned char* __restrict__ mask, int* __restrict__ absIdx, int B) {
  int b = blockIdx.x * blockDim.x + threadIdx.x;
  if (b >= B) return;
  int a = 0;
  for (int i = 0; i < PER_B; ++i) { int n = b * PER_B + i; if (mask[n]) a += n; }
  absIdx[b] = a;
}

// ---- segment softmax attention (batch segments are contiguous blocks of 50)
__global__ void attn_kernel(const float* __restrict__ kk, const float* __restrict__ vv,
                            const float* __restrict__ q, float* __restrict__ ctx) {
  __shared__ float qv[64];
  __shared__ float lg[PER_B];
  int b = blockIdx.x, t = threadIdx.x;
  qv[t] = q[(size_t)b * 64 + t];
  __syncthreads();
  for (int i = t; i < PER_B; i += 64) {
    int n = b * PER_B + i;
    float d = 0.f;
    for (int j = 0; j < 64; ++j) d += kk[(size_t)n * 64 + j] * qv[j];
    lg[i] = d * 0.125f;
  }
  __syncthreads();
  float mx = -1e30f;
  for (int i = 0; i < PER_B; ++i) mx = fmaxf(mx, lg[i]);
  float den = 0.f;
  for (int i = 0; i < PER_B; ++i) den += expf(lg[i] - mx);
  float acc = 0.f;
  for (int i = 0; i < PER_B; ++i)
    acc += (expf(lg[i] - mx) / den) * vv[((size_t)(b * PER_B + i)) * 64 + t];
  ctx[(size_t)b * 64 + t] = acc;
}

__global__ void zread_kernel(const float* __restrict__ sfin, const float* __restrict__ vfin,
                             const float* __restrict__ tfin, const float* __restrict__ ctx,
                             const int* __restrict__ absIdx, float* __restrict__ zr, int B) {
  int idx = blockIdx.x * blockDim.x + threadIdx.x;
  if (idx >= B * 192) return;
  int b = idx / 192, j = idx % 192;
  int a = absIdx[b];
  float v = 0.f;
  if (j < 64) v = sfin[(size_t)a * 64 + j];
  else if (j < 128) v = ctx[(size_t)b * 64 + (j - 64)];
  else if (j < 160) { int k = j - 128; for (int c = 0; c < 3; ++c) { float x = vfin[((size_t)a * 32 + k) * 3 + c]; v += x * x; } }
  else if (j < 176) { int k = j - 160; for (int c = 0; c < 5; ++c) { float x = tfin[((size_t)a * 16 + k) * 5 + c]; v += x * x; } }
  zr[(size_t)b * 192 + j] = v;
}

extern "C" void kernel_launch(void* const* d_in, const int* in_sizes, int n_in,
                              void* d_out, int out_size, void* d_ws, size_t ws_size,
                              hipStream_t stream) {
  const int*   z      = (const int*)d_in[0];
  const float* pos    = (const float*)d_in[1];
  const int*   ei     = (const int*)d_in[2];
  const float* eshift = (const float*)d_in[3];
  const unsigned char* amask = (const unsigned char*)d_in[5];
  const float* emb = (const float*)d_in[6];
  const float* rw1 = (const float*)d_in[7];
  const float* rb1 = (const float*)d_in[8];
  const float* rw2 = (const float*)d_in[9];
  const float* rb2 = (const float*)d_in[10];
  const float* wp0 = (const float*)d_in[11];
  const float* wp1 = (const float*)d_in[12];
  const float* wp2 = (const float*)d_in[13];
  const float* s0w = (const float*)d_in[14];
  const float* s1w = (const float*)d_in[15];
  const float* s2w = (const float*)d_in[16];
  const float* wq  = (const float*)d_in[17];
  const float* wk  = (const float*)d_in[18];
  const float* wv  = (const float*)d_in[19];
  const float* m1  = (const float*)d_in[20];
  const float* mb1 = (const float*)d_in[21];
  const float* m2  = (const float*)d_in[22];
  const float* mb2 = (const float*)d_in[23];
  float* out = (float*)d_out;

  const int N = in_sizes[0];
  const int E = in_sizes[2] / 2;
  const int B = N / PER_B;
  const int* srcI = ei;
  const int* dstI = ei + E;

  size_t cur = 0; char* base = (char*)d_ws;
  auto alloc = [&](size_t bytes) -> char* {
    char* p = base + cur; cur += (bytes + 255) & ~(size_t)255; return p;
  };
  auto cdiv = [](long long a, long long b) { return (unsigned)((a + b - 1) / b); };

  // packed weights (tile = 512 halves = 1KB)
  _Float16* rw1p = (_Float16*)alloc((size_t)NLAYERS_C * 4 * 1024);
  _Float16* rw2p = (_Float16*)alloc((size_t)NLAYERS_C * 14 * 1024);
  _Float16* Wp   = (_Float16*)alloc((size_t)NLAYERS_C * 14 * 1024);
  _Float16* s0p  = (_Float16*)alloc((size_t)NLAYERS_C * 8 * 1024);
  _Float16* wqp  = (_Float16*)alloc(8 * 1024);
  _Float16* wkp  = (_Float16*)alloc(8 * 1024);
  _Float16* wvp  = (_Float16*)alloc(8 * 1024);
  _Float16* m1p  = (_Float16*)alloc(48 * 1024);
  _Float16* m2p  = (_Float16*)alloc(32 * 1024);
  _Float16* rbas = (_Float16*)alloc((size_t)E * 32 * 2);
  float* sh = (float*)alloc((size_t)E * 9 * 4);
  float* sbuf[2] = { (float*)alloc((size_t)N * 64 * 4), (float*)alloc((size_t)N * 64 * 4) };
  float* vbuf[2] = { (float*)alloc((size_t)N * 96 * 4), (float*)alloc((size_t)N * 96 * 4) };
  float* tbuf[2] = { (float*)alloc((size_t)N * 80 * 4), (float*)alloc((size_t)N * 80 * 4) };
  float* a0  = (float*)alloc((size_t)N * 64 * 4);
  float* av  = (float*)alloc((size_t)N * 96 * 4);
  float* at  = (float*)alloc((size_t)N * 80 * 4);
  float* kkb = (float*)alloc((size_t)N * 64 * 4);
  float* vvb = (float*)alloc((size_t)N * 64 * 4);
  float* qb  = (float*)alloc((size_t)B * 64 * 4);
  int*   absIdx = (int*)alloc((size_t)B * 4);
  float* ctx = (float*)alloc((size_t)B * 64 * 4);
  float* zr  = (float*)alloc((size_t)B * 192 * 4);
  float* hid = (float*)alloc((size_t)B * 128 * 4);
  (void)ws_size; (void)n_in; (void)out_size; (void)eshift;

  // ---- pack all weights into WMMA B-fragment order ----
  for (int L = 0; L < NLAYERS_C; ++L) {
    pack_b_kernel<<<cdiv(1 * 32 * 64, 256), 256, 0, stream>>>(
        rw1 + (size_t)L * NRAD * 64, NRAD, 64, 64, rw1p + (size_t)L * 4 * 512, 4, 0, 1);
    pack_b_kernel<<<cdiv(2 * 32 * 112, 256), 256, 0, stream>>>(
        rw2 + (size_t)L * 64 * 112, 64, 112, 112, rw2p + (size_t)L * 14 * 512, 7, 0, 2);
    pack_b_kernel<<<cdiv(2 * 32 * 64, 256), 256, 0, stream>>>(
        wp0 + (size_t)L * 64 * 64, 64, 64, 64, Wp + (size_t)L * 14 * 512, 7, 0, 2);
    pack_b_kernel<<<cdiv(2 * 32 * 32, 256), 256, 0, stream>>>(
        wp1 + (size_t)L * 64 * 32, 64, 32, 32, Wp + (size_t)L * 14 * 512, 7, 4, 2);
    pack_b_kernel<<<cdiv(2 * 32 * 16, 256), 256, 0, stream>>>(
        wp2 + (size_t)L * 64 * 16, 64, 16, 16, Wp + (size_t)L * 14 * 512, 7, 6, 2);
    pack_b_kernel<<<cdiv(2 * 32 * 64, 256), 256, 0, stream>>>(
        s0w + (size_t)L * 64 * 64, 64, 64, 64, s0p + (size_t)L * 8 * 512, 4, 0, 2);
  }
  pack_b_kernel<<<cdiv(2 * 32 * 64, 256), 256, 0, stream>>>(wq, 64, 64, 64, wqp, 4, 0, 2);
  pack_b_kernel<<<cdiv(2 * 32 * 64, 256), 256, 0, stream>>>(wk, 64, 64, 64, wkp, 4, 0, 2);
  pack_b_kernel<<<cdiv(2 * 32 * 64, 256), 256, 0, stream>>>(wv, 64, 64, 64, wvp, 4, 0, 2);
  pack_b_kernel<<<cdiv(6 * 32 * 128, 256), 256, 0, stream>>>(m1, 176, 128, 128, m1p, 8, 0, 6);
  pack_b_kernel<<<cdiv(4 * 32 * 128, 256), 256, 0, stream>>>(m2, 128, 128, 128, m2p, 8, 0, 4);

  // ---- geometry + init ----
  edge_geom_kernel<<<cdiv(E, 256), 256, 0, stream>>>(ei, pos, eshift, sh, rbas, E);
  init_s_kernel<<<cdiv((long long)N * 64, 256), 256, 0, stream>>>(z, emb, sbuf[0], N);
  zero_kernel<<<cdiv((long long)N * 96, 256), 256, 0, stream>>>(vbuf[0], (long long)N * 96);
  zero_kernel<<<cdiv((long long)N * 80, 256), 256, 0, stream>>>(tbuf[0], (long long)N * 80);

  int in = 0;
  for (int L = 0; L < NLAYERS_C; ++L) {
    int outI = in ^ 1;
    zero_kernel<<<cdiv((long long)N * 64, 256), 256, 0, stream>>>(a0, (long long)N * 64);
    zero_kernel<<<cdiv((long long)N * 96, 256), 256, 0, stream>>>(av, (long long)N * 96);
    zero_kernel<<<cdiv((long long)N * 80, 256), 256, 0, stream>>>(at, (long long)N * 80);

    edge_msg_kernel<<<cdiv((long long)((E + 15) / 16), 8), 256, 0, stream>>>(
        srcI, dstI, rbas, sh, sbuf[in],
        rw1p + (size_t)L * 4 * 512, rb1 + (size_t)L * 64,
        rw2p + (size_t)L * 14 * 512, rb2 + (size_t)L * 112,
        Wp + (size_t)L * 14 * 512, a0, av, at, E);

    // s_out = silu(s @ s0 + a0)
    wmma_gemm_kernel<<<cdiv((long long)((N + 15) / 16), 8), 256, 0, stream>>>(
        sbuf[in], nullptr, N, 64, 64, s0p + (size_t)L * 8 * 512, 2, 4,
        nullptr, a0, 1, sbuf[outI], 64);
    vt_update_kernel<<<cdiv((long long)N * 176, 256), 256, 0, stream>>>(
        vbuf[in], tbuf[in], s1w + (size_t)L * 32 * 32, s2w + (size_t)L * 16 * 16,
        av, at, vbuf[outI], tbuf[outI], N);
    in = outI;
  }
  float* sfin = sbuf[in]; float* vfin = vbuf[in]; float* tfin = tbuf[in];

  // ---- readout ----
  wmma_gemm_kernel<<<cdiv((long long)((N + 15) / 16), 8), 256, 0, stream>>>(
      sfin, nullptr, N, 64, 64, wkp, 2, 4, nullptr, nullptr, 0, kkb, 64);
  wmma_gemm_kernel<<<cdiv((long long)((N + 15) / 16), 8), 256, 0, stream>>>(
      sfin, nullptr, N, 64, 64, wvp, 2, 4, nullptr, nullptr, 0, vvb, 64);
  absidx_kernel<<<cdiv(B, 256), 256, 0, stream>>>(amask, absIdx, B);
  wmma_gemm_kernel<<<cdiv((long long)((B + 15) / 16), 8), 256, 0, stream>>>(
      sfin, absIdx, B, 64, 64, wqp, 2, 4, nullptr, nullptr, 0, qb, 64);
  attn_kernel<<<B, 64, 0, stream>>>(kkb, vvb, qb, ctx);
  zread_kernel<<<cdiv((long long)B * 192, 256), 256, 0, stream>>>(
      sfin, vfin, tfin, ctx, absIdx, zr, B);
  wmma_gemm_kernel<<<cdiv((long long)((B + 15) / 16), 8), 256, 0, stream>>>(
      zr, nullptr, B, 176, 192, m1p, 6, 8, mb1, nullptr, 1, hid, 128);
  wmma_gemm_kernel<<<cdiv((long long)((B + 15) / 16), 8), 256, 0, stream>>>(
      hid, nullptr, B, 128, 128, m2p, 4, 8, mb2, nullptr, 0, out, 128);
}
